// GroupedQueryAttention_84559316123929
// MI455X (gfx1250) — matrix operand order repaired and checked by reading, not compiled, not run on hardware
//
#include <hip/hip_runtime.h>

typedef __attribute__((ext_vector_type(8)))  __bf16 v8bf;
typedef __attribute__((ext_vector_type(16))) __bf16 v16bf;
typedef __attribute__((ext_vector_type(8)))  float  v8f;

#define E_DIM   2048
#define HQ_N    16
#define G_SZ    4
#define HKV_N   4
#define D_HEAD  128
#define KVD_N   512
#define B_SZ    2
#define N_SEQ   2048
#define M_TOT   (B_SZ * N_SEQ)

// ---------- helpers ----------

__device__ __forceinline__ __bf16 f2bf(float f) {
  union { float f; unsigned u; } a; a.f = f;
  unsigned u = a.u;
  u += 0x7fffu + ((u >> 16) & 1u);           // round-to-nearest-even
  union { unsigned short s; __bf16 b; } r; r.s = (unsigned short)(u >> 16);
  return r.b;
}

__device__ __forceinline__ v16bf combine16(v8bf lo, v8bf hi) {
  v16bf r;
#pragma unroll
  for (int i = 0; i < 8; ++i) { r[i] = lo[i]; r[i + 8] = hi[i]; }
  return r;
}

// A-fragment (16-bit, 16x32): lane holds row (lane&15); halves K = kk..kk+7
// and kk+16..kk+23 with kk = 8*(lane>>4). p points at the kk element.
__device__ __forceinline__ v16bf load_frag_a(const __bf16* p) {
  v8bf lo = *(const v8bf*)p;
  v8bf hi = *(const v8bf*)(p + 16);
  return combine16(lo, hi);
}

// B-fragment (16-bit, 32x16): lane holds column (lane&15); halves K = kk..kk+7
// and kk+16..kk+23 with kk = 8*(lane>>4). p points at the kk element.
__device__ __forceinline__ v16bf load_frag_b(const __bf16* p) {
  v8bf lo = *(const v8bf*)p;
  v8bf hi = *(const v8bf*)(p + 16);
  return combine16(lo, hi);
}

// CDNA5 async global->LDS copy (16B per lane), tracked by ASYNCcnt.
__device__ __forceinline__ void async_b128(unsigned lds_off, const void* gaddr) {
  asm volatile("global_load_async_to_lds_b128 %0, %1, off"
               :: "v"(lds_off), "v"((unsigned long long)(uintptr_t)gaddr)
               : "memory");
}
__device__ __forceinline__ void wait_async0() {
  asm volatile("s_wait_asynccnt 0x0" ::: "memory");
}
// LDS byte offset of a shared-memory pointer (flat LDS aperture keeps the
// group-segment offset in bits [31:0]).
__device__ __forceinline__ unsigned lds_off_of(const void* p) {
  return (unsigned)(uintptr_t)p;
}

// ---------- fp32 -> bf16 conversion ----------

__global__ void cvt_f32_bf16(const float* __restrict__ in,
                             __bf16* __restrict__ out, int n8) {
  int i = blockIdx.x * blockDim.x + threadIdx.x;
  int stride = gridDim.x * blockDim.x;
  for (; i < n8; i += stride) {
    const float4* p = (const float4*)in + 2 * (size_t)i;
    float4 a = p[0], b = p[1];
    v8bf o;
    o[0] = f2bf(a.x); o[1] = f2bf(a.y); o[2] = f2bf(a.z); o[3] = f2bf(a.w);
    o[4] = f2bf(b.x); o[5] = f2bf(b.y); o[6] = f2bf(b.z); o[7] = f2bf(b.w);
    *(v8bf*)(out + 8 * (size_t)i) = o;
  }
}

// ---------- bf16 GEMM: C[M,N] = X[M,K] @ W[N,K]^T ----------
// 128x128 block tile, BK=32, 128 threads = 4 waves (2x2), each wave computes
// a 64x64 subtile (4x4 WMMA tiles -> 16 WMMA per K-step vs 16 ds_load_b128).
// Tiles stream in with global_load_async_to_lds_b128, double-buffered.

#define LDA  40                 // padded LDS stride in halves (80B)
#define ABUF (128 * LDA)        // halves per buffer

template <bool OUT_F32>
__global__ __launch_bounds__(128) void gemm_xwt(
    const __bf16* __restrict__ X, const __bf16* __restrict__ W,
    float* __restrict__ Cf, __bf16* __restrict__ Cb,
    int M, int Nout, int K) {
  __shared__ __align__(16) __bf16 As[2 * ABUF];
  __shared__ __align__(16) __bf16 Bs[2 * ABUF];

  const int tid  = threadIdx.x;
  const int lane = tid & 31;
  const int w    = tid >> 5;        // 0..3
  const int ln   = lane & 15;
  const int lh   = lane >> 4;
  const int m0   = blockIdx.y * 128;
  const int n0   = blockIdx.x * 128;
  const int wr   = (w >> 1) * 64;
  const int wc   = (w & 1) * 64;

  // loader: thread t owns row t of both tiles (32 halves = 4 x b128)
  const __bf16* gA = X + (size_t)(m0 + tid) * K;
  const __bf16* gB = W + (size_t)(n0 + tid) * K;
  const unsigned ldsA = lds_off_of(As) + tid * (LDA * 2);
  const unsigned ldsB = lds_off_of(Bs) + tid * (LDA * 2);

  // prologue: stage k0 = 0 into buffer 0
#pragma unroll
  for (int i = 0; i < 4; ++i) {
    async_b128(ldsA + i * 16, gA + i * 8);
    async_b128(ldsB + i * 16, gB + i * 8);
  }

  v8f acc[4][4] = {};

  int buf = 0;
  for (int k0 = 0; k0 < K; k0 += 32, buf ^= 1) {
    wait_async0();
    __syncthreads();                 // tiles ready; prior reads complete
    if (k0 + 32 < K) {               // prefetch next K-slice into other buffer
      const unsigned nb = (unsigned)(buf ^ 1) * (ABUF * 2);
#pragma unroll
      for (int i = 0; i < 4; ++i) {
        async_b128(ldsA + nb + i * 16, gA + k0 + 32 + i * 8);
        async_b128(ldsB + nb + i * 16, gB + k0 + 32 + i * 8);
      }
    }
    const __bf16* Ab = As + buf * ABUF;
    const __bf16* Bb = Bs + buf * ABUF;
    v16bf a[4], bfr[4];
#pragma unroll
    for (int i = 0; i < 4; ++i)
      a[i] = load_frag_a(Ab + (wr + i * 16 + ln) * LDA + 8 * lh);
#pragma unroll
    for (int j = 0; j < 4; ++j)
      bfr[j] = load_frag_b(Bb + (wc + j * 16 + ln) * LDA + 8 * lh);
#pragma unroll
    for (int i = 0; i < 4; ++i)
#pragma unroll
      for (int j = 0; j < 4; ++j)
        acc[i][j] = __builtin_amdgcn_wmma_f32_16x16x32_bf16(
            false, a[i], false, bfr[j], (short)0, acc[i][j], false, false);
  }

  // Epilogue: C layout — VGPR e: row = e + 8*(lane>>4), col = lane&15.
#pragma unroll
  for (int i = 0; i < 4; ++i)
#pragma unroll
    for (int j = 0; j < 4; ++j) {
      int row0 = m0 + wr + i * 16 + 8 * lh;
      int col  = n0 + wc + j * 16 + ln;
#pragma unroll
      for (int e = 0; e < 8; ++e) {
        float v = acc[i][j][e];
        if (OUT_F32) Cf[(size_t)(row0 + e) * Nout + col] = v;
        else         Cb[(size_t)(row0 + e) * Nout + col] = f2bf(v);
      }
    }
}

// ---------- flash GQA attention ----------
// Grid: (N/128, HQ, B). Block = 256 thr = 8 waves; wave w owns 16 query rows.
// Streams over s in chunks of 64; K chunk async-DMA'd row-major into LDS,
// V chunk stored transposed (Vt[d][s]) so P@V B-fragments are contiguous.

#define LDK 136  // K-chunk LDS stride (halves)
#define LDV 72   // Vt / P LDS stride (halves)

__global__ __launch_bounds__(256) void gqa_attn(
    const __bf16* __restrict__ Q, const __bf16* __restrict__ Km,
    const __bf16* __restrict__ Vm, __bf16* __restrict__ O) {
  __shared__ __align__(16) __bf16 Ks[64 * LDK];
  __shared__ __align__(16) __bf16 Vt[128 * LDV];
  __shared__ __align__(16) __bf16 Ps[8 * 16 * LDV];

  const int tid  = threadIdx.x;
  const int lane = tid & 31;
  const int w    = tid >> 5;
  const int ln   = lane & 15;
  const int lh   = lane >> 4;
  const int b    = blockIdx.z;
  const int hq   = blockIdx.y;
  const int hkv  = hq >> 2;              // G = 4
  const int nbase = blockIdx.x * 128 + w * 16;

  const float scale = 0.0883883476483184f; // 1/sqrt(128)

  // Q A-fragments: 4 K-chunks of 32 across D=128, straight from global.
  v16bf qf[4];
  {
    const __bf16* qrow =
        Q + (size_t)(b * N_SEQ + nbase + ln) * E_DIM + hq * D_HEAD;
#pragma unroll
    for (int c = 0; c < 4; ++c) qf[c] = load_frag_a(qrow + c * 32 + 8 * lh);
  }

  v8f oacc[8] = {};
  float mrow[8], lsum[8];
#pragma unroll
  for (int j = 0; j < 8; ++j) { mrow[j] = -3.0e38f; lsum[j] = 0.f; }

  __bf16* Pw = Ps + w * (16 * LDV);

  const int lrow = tid >> 2;          // 0..63 (s within chunk)
  const int lc0  = (tid & 3) * 32;    // 0..96 (d offset)
  const unsigned ldsK = lds_off_of(Ks) + (lrow * LDK + lc0) * 2;

  for (int s0 = 0; s0 < N_SEQ; s0 += 64) {
    __syncthreads();
    {
      const __bf16* kg =
          Km + (size_t)(b * N_SEQ + s0 + lrow) * KVD_N + hkv * D_HEAD + lc0;
#pragma unroll
      for (int i = 0; i < 4; ++i) async_b128(ldsK + i * 16, kg + i * 8);

      const __bf16* vg =
          Vm + (size_t)(b * N_SEQ + s0 + lrow) * KVD_N + hkv * D_HEAD + lc0;
      v8bf vv[4];
#pragma unroll
      for (int q4 = 0; q4 < 4; ++q4) vv[q4] = ((const v8bf*)vg)[q4];
#pragma unroll
      for (int q4 = 0; q4 < 4; ++q4)
#pragma unroll
        for (int e = 0; e < 8; ++e)
          Vt[(lc0 + q4 * 8 + e) * LDV + lrow] = vv[q4][e];
    }
    wait_async0();
    __syncthreads();

    // S = Q K^T for 4 s-tiles of 16
    v8f st[4] = {};
#pragma unroll
    for (int t = 0; t < 4; ++t)
#pragma unroll
      for (int c = 0; c < 4; ++c) {
        v16bf kb = load_frag_b(Ks + (t * 16 + ln) * LDK + c * 32 + 8 * lh);
        st[t] = __builtin_amdgcn_wmma_f32_16x16x32_bf16(
            false, qf[c], false, kb, (short)0, st[t], false, false);
      }

    // scale, row max (reduce across 16-lane half), online rescale
    float mnew[8];
#pragma unroll
    for (int j = 0; j < 8; ++j) {
      float mx = -3.0e38f;
#pragma unroll
      for (int t = 0; t < 4; ++t) { st[t][j] *= scale; mx = fmaxf(mx, st[t][j]); }
      mx = fmaxf(mx, __shfl_xor(mx, 1));
      mx = fmaxf(mx, __shfl_xor(mx, 2));
      mx = fmaxf(mx, __shfl_xor(mx, 4));
      mx = fmaxf(mx, __shfl_xor(mx, 8));
      mnew[j] = fmaxf(mrow[j], mx);
    }
#pragma unroll
    for (int j = 0; j < 8; ++j) {
      float alpha = __expf(mrow[j] - mnew[j]);
      lsum[j] *= alpha;
#pragma unroll
      for (int dt = 0; dt < 8; ++dt) oacc[dt][j] *= alpha;
      mrow[j] = mnew[j];
    }

    // P = exp(S - m); row sums; stash bf16 P in per-wave LDS strip
#pragma unroll
    for (int j = 0; j < 8; ++j) {
      float ps = 0.f;
#pragma unroll
      for (int t = 0; t < 4; ++t) {
        float p = __expf(st[t][j] - mnew[j]);
        ps += p;
        Pw[(j + 8 * lh) * LDV + t * 16 + ln] = f2bf(p);
      }
      ps += __shfl_xor(ps, 1);
      ps += __shfl_xor(ps, 2);
      ps += __shfl_xor(ps, 4);
      ps += __shfl_xor(ps, 8);
      lsum[j] += ps;
    }

    // O += P @ V  (A = P from LDS, B = Vt rows)
    v16bf pa0 = load_frag_a(Pw + ln * LDV + 8 * lh);
    v16bf pa1 = load_frag_a(Pw + ln * LDV + 32 + 8 * lh);
#pragma unroll
    for (int dt = 0; dt < 8; ++dt) {
      v16bf vb0 = load_frag_b(Vt + (dt * 16 + ln) * LDV + 8 * lh);
      v16bf vb1 = load_frag_b(Vt + (dt * 16 + ln) * LDV + 32 + 8 * lh);
      oacc[dt] = __builtin_amdgcn_wmma_f32_16x16x32_bf16(
          false, pa0, false, vb0, (short)0, oacc[dt], false, false);
      oacc[dt] = __builtin_amdgcn_wmma_f32_16x16x32_bf16(
          false, pa1, false, vb1, (short)0, oacc[dt], false, false);
    }
  }

  // finalize: O /= l, write bf16 at column hq*D
#pragma unroll
  for (int j = 0; j < 8; ++j) {
    float inv = 1.0f / lsum[j];
    int n = nbase + j + 8 * lh;
    __bf16* orow = O + (size_t)(b * N_SEQ + n) * E_DIM + hq * D_HEAD;
#pragma unroll
    for (int dt = 0; dt < 8; ++dt)
      orow[dt * 16 + ln] = f2bf(oacc[dt][j] * inv);
  }
}

// ---------- launch ----------

extern "C" void kernel_launch(void* const* d_in, const int* in_sizes, int n_in,
                              void* d_out, int out_size, void* d_ws,
                              size_t ws_size, hipStream_t stream) {
  (void)in_sizes; (void)n_in; (void)out_size; (void)ws_size;

  const float* x  = (const float*)d_in[0];
  const float* Wq = (const float*)d_in[1];
  const float* Wk = (const float*)d_in[2];
  const float* Wv = (const float*)d_in[3];
  const float* Wo = (const float*)d_in[4];
  float* out = (float*)d_out;

  __bf16* p  = (__bf16*)d_ws;
  __bf16* xb = p; p += (size_t)M_TOT * E_DIM;
  __bf16* wq = p; p += (size_t)E_DIM * E_DIM;
  __bf16* wk = p; p += (size_t)KVD_N * E_DIM;
  __bf16* wv = p; p += (size_t)KVD_N * E_DIM;
  __bf16* wo = p; p += (size_t)E_DIM * E_DIM;
  __bf16* qb = p; p += (size_t)M_TOT * E_DIM;
  __bf16* kb = p; p += (size_t)M_TOT * KVD_N;
  __bf16* vb = p; p += (size_t)M_TOT * KVD_N;
  __bf16* ob = p; p += (size_t)M_TOT * E_DIM;

  auto cvt = [&](const float* src, __bf16* dst, size_t n) {
    int n8 = (int)(n / 8);
    int blocks = (n8 + 255) / 256;
    if (blocks > 2048) blocks = 2048;
    cvt_f32_bf16<<<blocks, 256, 0, stream>>>(src, dst, n8);
  };
  cvt(x,  xb, (size_t)M_TOT * E_DIM);
  cvt(Wq, wq, (size_t)E_DIM * E_DIM);
  cvt(Wk, wk, (size_t)KVD_N * E_DIM);
  cvt(Wv, wv, (size_t)KVD_N * E_DIM);
  cvt(Wo, wo, (size_t)E_DIM * E_DIM);

  dim3 blk(128);
  gemm_xwt<false><<<dim3(E_DIM / 128, M_TOT / 128), blk, 0, stream>>>(
      xb, wq, nullptr, qb, M_TOT, E_DIM, E_DIM);
  gemm_xwt<false><<<dim3(KVD_N / 128, M_TOT / 128), blk, 0, stream>>>(
      xb, wk, nullptr, kb, M_TOT, KVD_N, E_DIM);
  gemm_xwt<false><<<dim3(KVD_N / 128, M_TOT / 128), blk, 0, stream>>>(
      xb, wv, nullptr, vb, M_TOT, KVD_N, E_DIM);

  gqa_attn<<<dim3(N_SEQ / 128, HQ_N, B_SZ), dim3(256), 0, stream>>>(
      qb, kb, vb, ob);

  gemm_xwt<true><<<dim3(E_DIM / 128, M_TOT / 128), blk, 0, stream>>>(
      ob, wo, out, nullptr, M_TOT, E_DIM, E_DIM);
}
